// Decoder_28827820490792
// MI455X (gfx1250) — matrix operand order repaired and checked by reading, not compiled, not run on hardware
//
#include <hip/hip_runtime.h>
#include <stdint.h>

// ---------------------------------------------------------------------------
// Problem constants (match the JAX reference)
// ---------------------------------------------------------------------------
#define B_   32
#define N_   32
#define E_   64
#define HW_  1600
#define WIMG 40
#define D_   256
#define NH_  8
#define DH_  32
#define L_   6
#define Tn_  1024      // B*N
#define Te_  2048      // B*E
#define T_   3072      // Tn+Te

typedef __bf16 bf16_t;
typedef __bf16 v16bf __attribute__((ext_vector_type(16)));
typedef __bf16 bf16x8 __attribute__((ext_vector_type(8)));
typedef float  v8f   __attribute__((ext_vector_type(8)));
typedef int    v4i   __attribute__((ext_vector_type(4)));

// gfx1250 async global->LDS copy (ASYNCcnt-tracked). Guarded: falls back to a
// synchronous b128 copy if the builtin is not exposed by this toolchain.
// Probe result (round 3 stderr): builtin exists, param 1 is non-const
// addrspace(1) int4*, 4 args total.
#if defined(__HIP_DEVICE_COMPILE__) && __has_builtin(__builtin_amdgcn_global_load_async_to_lds_b128)
#define HAS_ASYNC_LDS 1
typedef __attribute__((address_space(1))) v4i as1_v4i;
typedef __attribute__((address_space(3))) v4i as3_v4i;
#else
#define HAS_ASYNC_LDS 0
#endif

// Convert 8 packed floats -> 8 bf16 (compiler emits v_cvt_pk_bf16_f32 pairs)
__device__ inline bf16x8 cvt8(float4 a, float4 b)
{
    bf16x8 r;
    r[0] = (bf16_t)a.x; r[1] = (bf16_t)a.y; r[2] = (bf16_t)a.z; r[3] = (bf16_t)a.w;
    r[4] = (bf16_t)b.x; r[5] = (bf16_t)b.y; r[6] = (bf16_t)b.z; r[7] = (bf16_t)b.w;
    return r;
}

// A-fragment (16x32, 16-bit): lane holds row = lane%16; elems e<8 -> K = e+8*hi,
// e>=8 -> K = 16+(e-8)+8*hi.  Both halves are contiguous runs of 8 -> two b128 loads.
__device__ inline v16bf ldA16(const bf16_t* row, int hi)
{
    bf16x8 x0 = *(const bf16x8*)(row + (hi << 3));
    bf16x8 x1 = *(const bf16x8*)(row + 16 + (hi << 3));
    return __builtin_shufflevector(x0, x1, 0,1,2,3,4,5,6,7,8,9,10,11,12,13,14,15);
}

// B-fragment (32x16, 16-bit): lane%16 = N, elems e<8 -> K = e+8*hi,
// e>=8 -> K = 16+(e-8)+8*hi; p points at (K=0, N), K stride = ks.
__device__ inline v16bf ldB16(const bf16_t* p, int ks, int hi)
{
    v16bf r;
    #pragma unroll
    for (int e = 0; e < 8; ++e) {
        r[e]     = p[(size_t)(e + (hi << 3)) * ks];
        r[e + 8] = p[(size_t)(16 + e + (hi << 3)) * ks];
    }
    return r;
}

// ---------------------------------------------------------------------------
// WMMA GEMM:  C[M,N] = act( A[M,K] @ op(W) + bias )
//   A f32 row-major.  WT=1: W stored N x K (use W^T).  WT=0: W stored K x N.
//   ACT: 1 = exact GELU.  OBF: write bf16.  OKT: bf16 output scattered into
//   [B][N][HW] layout (row = b*HW+hw -> C[b][col][hw]); used for K^T.
//   Block tile 64x64, K step 32, 4 waves, 2x2 WMMAs per wave per K step.
// ---------------------------------------------------------------------------
template<int WT>
__global__ __launch_bounds__(128)
void gemm_wmma_k(const float* __restrict__ A, const float* __restrict__ W,
                 const float* __restrict__ bias, void* __restrict__ C,
                 int M, int N, int K, int ACT, int OBF, int OKT)
{
    __shared__ __align__(16) bf16_t As[64][40];   // [m][k], 80B rows (16B mult)
    __shared__ __align__(16) bf16_t Bs[32][72];   // [k][n], 144B rows (16B mult)

    const int tid  = threadIdx.x;
    const int lane = tid & 31;
    const int wave = tid >> 5;
    const int mw   = (wave >> 1) << 5;   // 0 / 32
    const int nw   = (wave & 1) << 5;    // 0 / 32
    const int bm   = blockIdx.y * 64;
    const int bn   = blockIdx.x * 64;
    const int lo   = lane & 15;
    const int hi   = lane >> 4;

    v8f acc00 = {0.f,0.f,0.f,0.f,0.f,0.f,0.f,0.f};
    v8f acc01 = acc00, acc10 = acc00, acc11 = acc00;

    // staging coordinates (each thread moves 16 contiguous floats per tile)
    const int ar = tid >> 1,  ac = (tid & 1) * 16;   // A: 64 rows x 32 cols
    const int wn = tid >> 1,  wk = (tid & 1) * 16;   // WT=1: W rows (N) x K
    const int kk = tid >> 2,  nc = (tid & 3) * 16;   // WT=0: K rows x 64 cols

    for (int k0 = 0; k0 < K; k0 += 32) {
        {   // A tile
            const float4* ap = (const float4*)(A + (size_t)(bm + ar) * K + k0 + ac);
            float4 a0 = ap[0], a1 = ap[1], a2 = ap[2], a3 = ap[3];
            *(bf16x8*)&As[ar][ac]     = cvt8(a0, a1);
            *(bf16x8*)&As[ar][ac + 8] = cvt8(a2, a3);
        }
        if (WT) {   // W is N x K: load along K, transpose into Bs[k][n]
            const float4* wp = (const float4*)(W + (size_t)(bn + wn) * K + k0 + wk);
            float4 w0 = wp[0], w1 = wp[1], w2 = wp[2], w3 = wp[3];
            bf16x8 lo8 = cvt8(w0, w1), hi8 = cvt8(w2, w3);
            #pragma unroll
            for (int i = 0; i < 8; ++i) Bs[wk + i][wn]     = lo8[i];
            #pragma unroll
            for (int i = 0; i < 8; ++i) Bs[wk + 8 + i][wn] = hi8[i];
        } else {    // W is K x N: rows are contiguous in N
            const float4* wp = (const float4*)(W + (size_t)(k0 + kk) * N + bn + nc);
            float4 w0 = wp[0], w1 = wp[1], w2 = wp[2], w3 = wp[3];
            *(bf16x8*)&Bs[kk][nc]     = cvt8(w0, w1);
            *(bf16x8*)&Bs[kk][nc + 8] = cvt8(w2, w3);
        }
        if (k0 + 32 < K) {   // prefetch next K-step tiles
            __builtin_prefetch(A + (size_t)(bm + ar) * K + (k0 + 32) + ac, 0, 1);
            if (WT) __builtin_prefetch(W + (size_t)(bn + wn) * K + (k0 + 32) + wk, 0, 1);
            else    __builtin_prefetch(W + (size_t)(k0 + 32 + kk) * N + bn + nc, 0, 1);
        }
        __syncthreads();

        v16bf a0f = ldA16(&As[mw + lo][0],      hi);
        v16bf a1f = ldA16(&As[mw + 16 + lo][0], hi);
        v16bf b0f = ldB16(&Bs[0][nw + lo], 72, hi);
        v16bf b1f = ldB16(&Bs[0][nw + 16 + lo], 72, hi);
        acc00 = __builtin_amdgcn_wmma_f32_16x16x32_bf16(false, a0f, false, b0f, (short)0, acc00, false, false);
        acc01 = __builtin_amdgcn_wmma_f32_16x16x32_bf16(false, a0f, false, b1f, (short)0, acc01, false, false);
        acc10 = __builtin_amdgcn_wmma_f32_16x16x32_bf16(false, a1f, false, b0f, (short)0, acc10, false, false);
        acc11 = __builtin_amdgcn_wmma_f32_16x16x32_bf16(false, a1f, false, b1f, (short)0, acc11, false, false);
        __syncthreads();
    }

    // epilogue: C tile rows r + 8*hi, col = lane%16  (per 16x16 sub-tile)
    #pragma unroll
    for (int tm = 0; tm < 2; ++tm) {
        #pragma unroll
        for (int tn = 0; tn < 2; ++tn) {
            const v8f& acc = tm ? (tn ? acc11 : acc10) : (tn ? acc01 : acc00);
            #pragma unroll
            for (int r = 0; r < 8; ++r) {
                int row = bm + mw + tm * 16 + r + (hi << 3);
                int col = bn + nw + tn * 16 + lo;
                float v = acc[r];
                if (bias) v += bias[col];
                if (ACT == 1) v = 0.5f * v * (1.0f + erff(v * 0.70710678118654752f));
                if (OBF) {
                    if (OKT) {   // transposed [B][N][HW] output (K^T for attention)
                        int bb = row / HW_;
                        int hw = row - bb * HW_;
                        ((bf16_t*)C)[((size_t)bb * N + col) * HW_ + hw] = (bf16_t)v;
                    } else {
                        ((bf16_t*)C)[(size_t)row * N + col] = (bf16_t)v;
                    }
                } else {
                    ((float*)C)[(size_t)row * N + col] = v;
                }
            }
        }
    }
}

// ---------------------------------------------------------------------------
// Flash attention: one workgroup per (head, batch). 6 waves, each owns a
// 16-query tile (queries stored flat: nodes [b*32..), edges [Tn+b*64..)).
// K is pre-transposed ([B][D][HW]) so S-fragments load straight from global
// with 2-byte loads, keys contiguous across lanes; V tile is copied global->LDS with ASYNC b128 loads that
// overlap the S WMMAs + online softmax; P is relaid via LDS into an A
// fragment; 2+2 WMMAs per 32-key step.
// ---------------------------------------------------------------------------
__global__ __launch_bounds__(192)
void flash_attn_k(const bf16_t* __restrict__ qh, const bf16_t* __restrict__ khT,
                  const bf16_t* __restrict__ vh, const float* __restrict__ bias,
                  float* __restrict__ ob)
{
    __shared__ __align__(16) bf16_t Vs[32][40];       // [key][dh]
    __shared__ __align__(16) bf16_t pbuf[6][16][40];  // per-wave P tile

    const int hh   = blockIdx.x;          // head
    const int b    = blockIdx.y;          // batch
    const int tid  = threadIdx.x;
    const int lane = tid & 31;
    const int qm   = tid >> 5;            // wave id = query tile 0..5
    const int lo   = lane & 15;
    const int hi   = lane >> 4;
    const int fr   = (qm < 2) ? (b * N_ + qm * 16)
                              : (Tn_ + b * E_ + (qm - 2) * 16);
    const float scale = 0.17677669529663687f;  // 1/sqrt(32)

    // Q fragment (16 queries x 32 dh) via two b128 global loads per lane
    v16bf qa = ldA16(qh + (size_t)(fr + lo) * D_ + hh * DH_, hi);
    // K^T base for this head: [dh][key], key stride 1, dh stride HW_
    const bf16_t* kbase = khT + ((size_t)b * D_ + hh * DH_) * HW_;

    v8f o0 = {0.f,0.f,0.f,0.f,0.f,0.f,0.f,0.f};
    v8f o1 = o0;
    float mrow[8], lrow[8];
    #pragma unroll
    for (int r = 0; r < 8; ++r) { mrow[r] = -1e30f; lrow[r] = 0.f; }

    for (int k0 = 0; k0 < HW_; k0 += 32) {
        // stage V [key][dh] into LDS (threads 0..127), async if available —
        // overlaps with the S WMMAs and softmax below.
        if (tid < 128) {
            int key = tid >> 2, c0 = (tid & 3) * 8;
            const bf16_t* src = vh + (size_t)(b * HW_ + k0 + key) * D_ + hh * DH_ + c0;
#if HAS_ASYNC_LDS
            __builtin_amdgcn_global_load_async_to_lds_b128(
                (as1_v4i*)(uintptr_t)src,
                (as3_v4i*)(uint32_t)(uintptr_t)&Vs[key][c0], 0, 0);
#else
            *(bf16x8*)&Vs[key][c0] = *(const bf16x8*)src;
#endif
        }

        // S = Q K^T  (B fragments straight from global: contiguous keys)
        v16bf kb0 = ldB16(kbase + k0 + lo, HW_, hi);
        v16bf kb1 = ldB16(kbase + k0 + 16 + lo, HW_, hi);
        v8f s0 = {0.f,0.f,0.f,0.f,0.f,0.f,0.f,0.f};
        v8f s1 = s0;
        s0 = __builtin_amdgcn_wmma_f32_16x16x32_bf16(false, qa, false, kb0, (short)0, s0, false, false);
        s1 = __builtin_amdgcn_wmma_f32_16x16x32_bf16(false, qa, false, kb1, (short)0, s1, false, false);

        // scale + bias, online softmax (row = r + 8*hi, col = lane%16)
        #pragma unroll
        for (int r = 0; r < 8; ++r) {
            int row = fr + r + (hi << 3);
            s0[r] = s0[r] * scale + bias[(size_t)row * HW_ + k0 + lo];
            s1[r] = s1[r] * scale + bias[(size_t)row * HW_ + k0 + 16 + lo];
            float mx = fmaxf(s0[r], s1[r]);
            #pragma unroll
            for (int m = 1; m <= 8; m <<= 1) mx = fmaxf(mx, __shfl_xor(mx, m, 32));
            float newm = fmaxf(mrow[r], mx);
            float corr = __expf(mrow[r] - newm);
            float p0 = __expf(s0[r] - newm);
            float p1 = __expf(s1[r] - newm);
            float rs = p0 + p1;
            #pragma unroll
            for (int m = 1; m <= 8; m <<= 1) rs += __shfl_xor(rs, m, 32);
            lrow[r] = lrow[r] * corr + rs;
            mrow[r] = newm;
            o0[r] *= corr; o1[r] *= corr;
            s0[r] = p0; s1[r] = p1;
        }
        #pragma unroll
        for (int r = 0; r < 8; ++r) {
            pbuf[qm][r + (hi << 3)][lo]      = (bf16_t)s0[r];
            pbuf[qm][r + (hi << 3)][lo + 16] = (bf16_t)s1[r];
        }
#if HAS_ASYNC_LDS
        asm volatile("s_wait_asynccnt 0x0" ::: "memory");
#endif
        __syncthreads();

        // O += P V  (P as A fragment, V as two B fragments, P via b128 LDS loads, V via strided 2-byte loads)
        v16bf pa  = ldA16(&pbuf[qm][lo][0], hi);
        v16bf vb0 = ldB16(&Vs[0][lo], 40, hi);
        v16bf vb1 = ldB16(&Vs[0][16 + lo], 40, hi);
        o0 = __builtin_amdgcn_wmma_f32_16x16x32_bf16(false, pa, false, vb0, (short)0, o0, false, false);
        o1 = __builtin_amdgcn_wmma_f32_16x16x32_bf16(false, pa, false, vb1, (short)0, o1, false, false);
        __syncthreads();   // protect Vs/pbuf before next-iteration staging
    }

    #pragma unroll
    for (int r = 0; r < 8; ++r) {
        int row = fr + r + (hi << 3);
        float inv = 1.0f / lrow[r];
        ob[(size_t)row * D_ + hh * DH_ + lo]      = o0[r] * inv;
        ob[(size_t)row * D_ + hh * DH_ + 16 + lo] = o1[r] * inv;
    }
}

// ---------------------------------------------------------------------------
// LayerNorm: one wave per 256-element row (block = 8 waves)
// ---------------------------------------------------------------------------
__global__ __launch_bounds__(256)
void layernorm_k(const float* __restrict__ in, const float* __restrict__ g,
                 const float* __restrict__ bt, float* __restrict__ out)
{
    int row  = blockIdx.x * 8 + (threadIdx.x >> 5);
    int lane = threadIdx.x & 31;
    const float* x = in + (size_t)row * D_;
    float s = 0.f, ss = 0.f;
    #pragma unroll
    for (int i = 0; i < 8; ++i) { float v = x[lane + i * 32]; s += v; ss += v * v; }
    #pragma unroll
    for (int m = 1; m <= 16; m <<= 1) { s += __shfl_xor(s, m, 32); ss += __shfl_xor(ss, m, 32); }
    float mean = s * (1.0f / 256.0f);
    float var  = ss * (1.0f / 256.0f) - mean * mean;
    float rstd = rsqrtf(var + 1e-5f);
    #pragma unroll
    for (int i = 0; i < 8; ++i) {
        int d = lane + i * 32;
        out[(size_t)row * D_ + d] = (x[d] - mean) * rstd * g[d] + bt[d];
    }
}

// ---------------------------------------------------------------------------
// Precompute kernels
// ---------------------------------------------------------------------------
__device__ inline float hm_val(float cx, float cy, float w, float h, int p)
{
    int py = p / WIMG, px = p % WIMG;
    float gx = (px + 0.5f) / 40.0f, gy = (py + 0.5f) / 40.0f;
    float dx = gx - cx, dy = gy - cy;
    float hx = __expf(-dx * dx / (2.0f * (0.5f * w) * (0.5f * w) + 1e-6f));
    float hy = __expf(-dy * dy / (2.0f * (0.5f * h) * (0.5f * h) + 1e-6f));
    return hy * hx;
}

__global__ void edge_prep_k(const int* __restrict__ el, const float* __restrict__ boxes,
                            int* __restrict__ srcdst, float* __restrict__ ubox,
                            float* __restrict__ delta)
{
    int e = blockIdx.x * blockDim.x + threadIdx.x;
    if (e >= Te_) return;
    int b = e / E_;
    int s = el[e] + b * N_;
    int d = el[Te_ + e] + b * N_;
    srcdst[e] = s; srcdst[Te_ + e] = d;
    float c1[4], c2[4];
    #pragma unroll
    for (int i = 0; i < 4; ++i) { c1[i] = boxes[s * 4 + i]; c2[i] = boxes[d * 4 + i]; }
    float x11 = c1[0] - 0.5f * c1[2], y11 = c1[1] - 0.5f * c1[3];
    float x12 = c1[0] + 0.5f * c1[2], y12 = c1[1] + 0.5f * c1[3];
    float x21 = c2[0] - 0.5f * c2[2], y21 = c2[1] - 0.5f * c2[3];
    float x22 = c2[0] + 0.5f * c2[2], y22 = c2[1] + 0.5f * c2[3];
    float ux1 = fminf(x11, x21), uy1 = fminf(y11, y21);
    float ux2 = fmaxf(x12, x22), uy2 = fmaxf(y12, y22);
    ubox[e * 4 + 0] = 0.5f * (ux1 + ux2);
    ubox[e * 4 + 1] = 0.5f * (uy1 + uy2);
    ubox[e * 4 + 2] = ux2 - ux1;
    ubox[e * 4 + 3] = uy2 - uy1;
    #pragma unroll
    for (int i = 0; i < 4; ++i) delta[e * 4 + i] = c1[i] - c2[i];
}

__global__ void node_hm_k(const float* __restrict__ boxes, float* __restrict__ bias)
{
    int idx = blockIdx.x * blockDim.x + threadIdx.x;
    if (idx >= Tn_ * HW_) return;
    int t = idx / HW_, p = idx % HW_;
    bias[idx] = hm_val(boxes[t * 4], boxes[t * 4 + 1], boxes[t * 4 + 2], boxes[t * 4 + 3], p);
}

__global__ void edge_hm_k(const float* __restrict__ ubox, const int* __restrict__ srcdst,
                          float* __restrict__ bias)
{
    int idx = blockIdx.x * blockDim.x + threadIdx.x;
    if (idx >= Te_ * HW_) return;
    int e = idx / HW_, p = idx % HW_;
    int s = srcdst[e], d = srcdst[Te_ + e];
    float v = fmaxf(bias[(size_t)s * HW_ + p], bias[(size_t)d * HW_ + p]);
    v = fmaxf(v, hm_val(ubox[e * 4], ubox[e * 4 + 1], ubox[e * 4 + 2], ubox[e * 4 + 3], p));
    bias[(size_t)(Tn_ + e) * HW_ + p] = v;
}

__global__ void emb_k(const float* __restrict__ boxes, const float* __restrict__ delta,
                      float* __restrict__ emb)
{
    int idx = blockIdx.x * blockDim.x + threadIdx.x;
    if (idx >= T_ * D_) return;
    int t = idx / D_, d = idx % D_;
    int i = d >> 6, j = d & 63, jj = j & 31;
    float v = (t < Tn_) ? boxes[t * 4 + i] : delta[(t - Tn_) * 4 + i];
    float f = __expf(-9.210340371976184f * (float)jj / 32.0f);  // exp(-ln(1e4)*jj/32)
    float ang = v * f;
    emb[idx] = (j < 32) ? __sinf(ang) : __cosf(ang);
}

__global__ void init_q_k(const float* __restrict__ n0, const float* __restrict__ e0,
                         float* __restrict__ q)
{
    int idx = blockIdx.x * blockDim.x + threadIdx.x;
    if (idx >= T_ * D_) return;
    q[idx] = (idx < Tn_ * D_) ? n0[idx] : e0[idx - Tn_ * D_];
}

__global__ void add_emb_k(const float* __restrict__ x, const float* __restrict__ emb,
                          float* __restrict__ xe)
{
    int idx = blockIdx.x * blockDim.x + threadIdx.x;
    if (idx >= T_ * D_) return;
    xe[idx] = x[idx] + emb[idx];
}

// ---------------------------------------------------------------------------
// GAT kernels
// ---------------------------------------------------------------------------
__global__ void gat_logits_k(const float* __restrict__ hbuf, const int* __restrict__ srcdst,
                             const float* __restrict__ a_src, const float* __restrict__ a_dst,
                             const float* __restrict__ a_edge, float* __restrict__ logits)
{
    int idx = blockIdx.x * blockDim.x + threadIdx.x;
    if (idx >= Te_ * NH_) return;
    int e = idx / NH_, hd = idx % NH_;
    int s = srcdst[e], d = srcdst[Te_ + e];
    float acc = 0.f;
    #pragma unroll
    for (int k = 0; k < DH_; ++k) {
        int c = hd * DH_ + k;
        acc += hbuf[(size_t)s * D_ + c] * a_src[c]
             + hbuf[(size_t)d * D_ + c] * a_dst[c]
             + hbuf[(size_t)(Tn_ + e) * D_ + c] * a_edge[c];
    }
    logits[idx] = (acc > 0.f) ? acc : 0.2f * acc;   // leaky_relu(0.2)
}

__global__ void gat_seg_k(const float* __restrict__ logits, const int* __restrict__ srcdst,
                          float* __restrict__ mbuf, float* __restrict__ den)
{
    int idx = blockIdx.x * blockDim.x + threadIdx.x;
    if (idx >= Tn_ * NH_) return;
    int t = idx / NH_, hd = idx % NH_;
    int b = t / N_;
    float m = -1e30f;
    for (int e2 = 0; e2 < E_; ++e2) {
        int e = b * E_ + e2;
        if (srcdst[Te_ + e] == t) m = fmaxf(m, logits[e * NH_ + hd]);
    }
    float s = 0.f;
    for (int e2 = 0; e2 < E_; ++e2) {
        int e = b * E_ + e2;
        if (srcdst[Te_ + e] == t) s += __expf(logits[e * NH_ + hd] - m);
    }
    mbuf[idx] = m; den[idx] = s;
}

__global__ void gat_aggr_k(const float* __restrict__ hbuf, const float* __restrict__ logits,
                           const int* __restrict__ srcdst, const float* __restrict__ mbuf,
                           const float* __restrict__ den, float* __restrict__ nout)
{
    int idx = blockIdx.x * blockDim.x + threadIdx.x;
    if (idx >= Tn_ * D_) return;
    int t = idx / D_, d = idx % D_;
    int hd = d / DH_, b = t / N_;
    float m  = mbuf[t * NH_ + hd];
    float dn = den[t * NH_ + hd] + 1e-9f;
    float acc = 0.f;
    for (int e2 = 0; e2 < E_; ++e2) {
        int e = b * E_ + e2;
        if (srcdst[Te_ + e] == t) {
            float a = __expf(logits[e * NH_ + hd] - m) / dn;
            acc += a * (hbuf[(size_t)srcdst[e] * D_ + d] + hbuf[(size_t)(Tn_ + e) * D_ + d]);
        }
    }
    nout[idx] = acc;
}

// ---------------------------------------------------------------------------
// Residual updates. mode 0: q += ls*y ; mode 1: q = base + ls*y.
// outdst (optional): also write q there (per-layer output capture).
// ---------------------------------------------------------------------------
__global__ void resid_k(float* __restrict__ q, const float* __restrict__ base,
                        const float* __restrict__ ls, const float* __restrict__ y,
                        float* __restrict__ outdst, int mode)
{
    int idx = blockIdx.x * blockDim.x + threadIdx.x;
    if (idx >= T_ * D_) return;
    int d = idx % D_;
    float v = (mode == 1) ? (base[idx] + ls[d] * y[idx]) : (q[idx] + ls[d] * y[idx]);
    q[idx] = v;
    if (outdst) outdst[idx] = v;
}

// ---------------------------------------------------------------------------
// Host driver
// ---------------------------------------------------------------------------
extern "C" void kernel_launch(void* const* d_in, const int* in_sizes, int n_in,
                              void* d_out, int out_size, void* d_ws, size_t ws_size,
                              hipStream_t stream)
{
    (void)in_sizes; (void)n_in; (void)out_size; (void)ws_size;
    const float* features   = (const float*)d_in[0];
    const float* boxes      = (const float*)d_in[1];
    const int*   edge_local = (const int*)  d_in[2];
    const float* nodes0     = (const float*)d_in[3];
    const float* edges0     = (const float*)d_in[4];
    const float* ln1_g      = (const float*)d_in[5];
    const float* ln1_b      = (const float*)d_in[6];
    const float* attn_in_w  = (const float*)d_in[7];
    const float* attn_in_b  = (const float*)d_in[8];
    const float* attn_out_w = (const float*)d_in[9];
    const float* attn_out_b = (const float*)d_in[10];
    const float* ls1        = (const float*)d_in[11];
    const float* ln2_g      = (const float*)d_in[12];
    const float* ln2_b      = (const float*)d_in[13];
    const float* gat_wn     = (const float*)d_in[14];
    const float* gat_we     = (const float*)d_in[15];
    const float* gat_a_src  = (const float*)d_in[16];
    const float* gat_a_dst  = (const float*)d_in[17];
    const float* gat_a_edge = (const float*)d_in[18];
    const float* gat_wo     = (const float*)d_in[19];
    const float* gat_bo     = (const float*)d_in[20];
    const float* gat_woe    = (const float*)d_in[21];
    const float* gat_boe    = (const float*)d_in[22];
    const float* ls2        = (const float*)d_in[23];
    const float* ln3_g      = (const float*)d_in[24];
    const float* ln3_b      = (const float*)d_in[25];
    const float* ffn_w1     = (const float*)d_in[26];
    const float* ffn_b1     = (const float*)d_in[27];
    const float* ffn_w2     = (const float*)d_in[28];
    const float* ffn_b2     = (const float*)d_in[29];
    const float* ls3        = (const float*)d_in[30];

    char* ws = (char*)d_ws;
    size_t off = 0;
    auto alloc = [&](size_t bytes) -> char* {
        char* p = ws + off;
        off = (off + bytes + 255) & ~(size_t)255;
        return p;
    };
    float*  bias   = (float*) alloc((size_t)T_ * HW_ * 4);
    int*    srcdst = (int*)   alloc((size_t)Te_ * 2 * 4);
    float*  ubox   = (float*) alloc((size_t)Te_ * 4 * 4);
    float*  delta  = (float*) alloc((size_t)Te_ * 4 * 4);
    float*  emb    = (float*) alloc((size_t)T_ * D_ * 4);
    float*  q      = (float*) alloc((size_t)T_ * D_ * 4);
    float*  xbuf   = (float*) alloc((size_t)T_ * D_ * 4);
    float*  xebuf  = (float*) alloc((size_t)T_ * D_ * 4);
    bf16_t* qh     = (bf16_t*)alloc((size_t)T_ * D_ * 2);
    bf16_t* khT    = (bf16_t*)alloc((size_t)B_ * HW_ * D_ * 2);
    bf16_t* vh     = (bf16_t*)alloc((size_t)B_ * HW_ * D_ * 2);
    float*  ob     = (float*) alloc((size_t)T_ * D_ * 4);
    float*  o2     = (float*) alloc((size_t)T_ * D_ * 4);
    float*  hbuf   = (float*) alloc((size_t)T_ * D_ * 4);
    float*  logits = (float*) alloc((size_t)Te_ * NH_ * 4);
    float*  mbuf   = (float*) alloc((size_t)Tn_ * NH_ * 4);
    float*  den    = (float*) alloc((size_t)Tn_ * NH_ * 4);
    float*  nout   = (float*) alloc((size_t)Tn_ * D_ * 4);
    float*  gatout = (float*) alloc((size_t)T_ * D_ * 4);
    float*  ffh    = (float*) alloc((size_t)T_ * 4 * D_ * 4);
    float*  y2     = (float*) alloc((size_t)T_ * D_ * 4);

    const int TPB = 256;
    // ---- precompute (layer-invariant) ----
    edge_prep_k<<<(Te_ + TPB - 1) / TPB, TPB, 0, stream>>>(edge_local, boxes, srcdst, ubox, delta);
    node_hm_k <<<(Tn_ * HW_ + TPB - 1) / TPB, TPB, 0, stream>>>(boxes, bias);
    edge_hm_k <<<(Te_ * HW_ + TPB - 1) / TPB, TPB, 0, stream>>>(ubox, srcdst, bias);
    emb_k     <<<(T_ * D_ + TPB - 1) / TPB, TPB, 0, stream>>>(boxes, delta, emb);
    init_q_k  <<<(T_ * D_ + TPB - 1) / TPB, TPB, 0, stream>>>(nodes0, edges0, q);

    const int nTD = (T_ * D_ + TPB - 1) / TPB;
    for (int l = 0; l < L_; ++l) {
        const float* Wq = attn_in_w + (size_t)l * 3 * D_ * D_;
        const float* Wk = Wq + (size_t)D_ * D_;
        const float* Wv = Wk + (size_t)D_ * D_;
        const float* bq = attn_in_b + (size_t)l * 3 * D_;
        const float* bk = bq + D_;
        const float* bv = bk + D_;

        // ---- cross attention ----
        layernorm_k<<<T_ / 8, 256, 0, stream>>>(q, ln1_g + l * D_, ln1_b + l * D_, xbuf);
        gemm_wmma_k<1><<<dim3(D_ / 64, T_ / 64), 128, 0, stream>>>(
            xbuf, Wq, bq, qh, T_, D_, D_, 0, 1, 0);
        gemm_wmma_k<1><<<dim3(D_ / 64, (B_ * HW_) / 64), 128, 0, stream>>>(
            features, Wk, bk, khT, B_ * HW_, D_, D_, 0, 1, 1);   // K^T layout
        gemm_wmma_k<1><<<dim3(D_ / 64, (B_ * HW_) / 64), 128, 0, stream>>>(
            features, Wv, bv, vh, B_ * HW_, D_, D_, 0, 1, 0);
        flash_attn_k<<<dim3(NH_, B_), 192, 0, stream>>>(qh, khT, vh, bias, ob);
        gemm_wmma_k<1><<<dim3(D_ / 64, T_ / 64), 128, 0, stream>>>(
            ob, attn_out_w + (size_t)l * D_ * D_, attn_out_b + l * D_, o2, T_, D_, D_, 0, 0, 0);
        resid_k<<<nTD, TPB, 0, stream>>>(q, nullptr, ls1 + l * D_, o2, nullptr, 0);

        // ---- GAT ----
        layernorm_k<<<T_ / 8, 256, 0, stream>>>(q, ln2_g + l * D_, ln2_b + l * D_, xbuf);
        add_emb_k<<<nTD, TPB, 0, stream>>>(xbuf, emb, xebuf);
        gemm_wmma_k<0><<<dim3(D_ / 64, Tn_ / 64), 128, 0, stream>>>(
            xebuf, gat_wn + (size_t)l * D_ * D_, nullptr, hbuf, Tn_, D_, D_, 0, 0, 0);
        gemm_wmma_k<0><<<dim3(D_ / 64, Te_ / 64), 128, 0, stream>>>(
            xebuf + (size_t)Tn_ * D_, gat_we + (size_t)l * D_ * D_, nullptr,
            hbuf + (size_t)Tn_ * D_, Te_, D_, D_, 0, 0, 0);
        gat_logits_k<<<(Te_ * NH_ + TPB - 1) / TPB, TPB, 0, stream>>>(
            hbuf, srcdst, gat_a_src + l * D_, gat_a_dst + l * D_, gat_a_edge + l * D_, logits);
        gat_seg_k<<<(Tn_ * NH_ + TPB - 1) / TPB, TPB, 0, stream>>>(logits, srcdst, mbuf, den);
        gat_aggr_k<<<(Tn_ * D_ + TPB - 1) / TPB, TPB, 0, stream>>>(
            hbuf, logits, srcdst, mbuf, den, nout);
        gemm_wmma_k<0><<<dim3(D_ / 64, Tn_ / 64), 128, 0, stream>>>(
            nout, gat_wo + (size_t)l * D_ * D_, gat_bo + l * D_, gatout, Tn_, D_, D_, 0, 0, 0);
        gemm_wmma_k<0><<<dim3(D_ / 64, Te_ / 64), 128, 0, stream>>>(
            hbuf + (size_t)Tn_ * D_, gat_woe + (size_t)l * D_ * D_, gat_boe + l * D_,
            gatout + (size_t)Tn_ * D_, Te_, D_, D_, 0, 0, 0);
        resid_k<<<nTD, TPB, 0, stream>>>(q, xbuf, ls2 + l * D_, gatout, nullptr, 1);

        // ---- FFN ----
        layernorm_k<<<T_ / 8, 256, 0, stream>>>(q, ln3_g + l * D_, ln3_b + l * D_, xbuf);
        gemm_wmma_k<0><<<dim3((4 * D_) / 64, T_ / 64), 128, 0, stream>>>(
            xbuf, ffn_w1 + (size_t)l * D_ * 4 * D_, ffn_b1 + (size_t)l * 4 * D_,
            ffh, T_, 4 * D_, D_, 1, 0, 0);
        gemm_wmma_k<0><<<dim3(D_ / 64, T_ / 64), 128, 0, stream>>>(
            ffh, ffn_w2 + (size_t)l * 4 * D_ * D_, ffn_b2 + l * D_, y2, T_, D_, 4 * D_, 0, 0, 0);
        resid_k<<<nTD, TPB, 0, stream>>>(q, nullptr, ls3 + l * D_, y2,
                                         (float*)d_out + (size_t)l * T_ * D_, 0);
    }
}
